// GPT_78709570666612
// MI455X (gfx1250) — matrix operand order repaired and checked by reading, not compiled, not run on hardware
//
#include <hip/hip_runtime.h>
#include <math.h>

// GPT-2 forward, MI455X (gfx1250), wave32 + v_wmma_f32_16x16x32_f16 everywhere.
#define TT 1024
#define DD 1024
#define NH 16
#define HDIM 64
#define NL 12
#define VV 50257

typedef __attribute__((ext_vector_type(16))) _Float16 v16h;
typedef __attribute__((ext_vector_type(8)))  float    v8f;

union FragH { v16h v; unsigned u[8]; };

__device__ __forceinline__ unsigned short f2h(float f) {
  _Float16 x = (_Float16)f;
  return __builtin_bit_cast(unsigned short, x);
}

// ISA 7.12.2: 16-bit A-matrix 16x32. Element e=(g,p): g<4 -> K=8*half+2g+p ; g>=4 -> K=16+8*half+2(g-4)+p
__device__ __forceinline__ int a_koff(int g, int hf) {
  return (g < 4) ? (8 * hf + 2 * g) : (16 + 8 * hf + 2 * (g - 4));
}

__device__ __forceinline__ float gelu_fn(float x) {
  float x3 = x * x * x;
  return 0.5f * x * (1.0f + tanhf(0.7978845608028654f * (x + 0.044715f * x3)));
}

// ---------------------------------------------------------------------------
// Tiled WMMA GEMM: C[M,N] = A[M,K] * B + bias (+resid) (gelu optional)
// TRANSB: B element (k,n) = Bm[n*K + k]  (LM head: logits = h @ lm_w^T)
// Block: 256 thr (8 waves), tile 64(M) x 128(N), K-step 32. Wave tile 32x32.
// ---------------------------------------------------------------------------
template <bool TRANSB, bool RESID, bool GELU>
__global__ __launch_bounds__(256) void gemm_kernel(
    const float* __restrict__ A, const float* __restrict__ Bm,
    const float* __restrict__ bias, const float* __restrict__ resid,
    float* __restrict__ C, int M, int N, int K) {
  __shared__ unsigned short As[64 * 40];   // [m][k], pad to 40
  __shared__ unsigned short Bs[128 * 40];  // [n][k], pad to 40

  const int tid = threadIdx.x;
  const int lane = tid & 31;
  const int wid = tid >> 5;
  const int hf = lane >> 4;
  const int m15 = lane & 15;
  const int wm = wid >> 2;  // 0..1
  const int wn = wid & 3;   // 0..3
  const int m0 = blockIdx.y * 64;
  const int n0 = blockIdx.x * 128;

  v8f acc[2][2] = {};

  const int arow = tid >> 2;         // 0..63
  const int acol = (tid & 3) << 3;   // 0,8,16,24

  for (int k0 = 0; k0 < K; k0 += 32) {
    {  // A tile 64x32 fp32 -> f16 LDS
      const float* src = A + (size_t)(m0 + arow) * K + k0 + acol;
      unsigned short* dst = &As[arow * 40 + acol];
#pragma unroll
      for (int i = 0; i < 8; ++i) dst[i] = f2h(src[i]);
    }
    if (!TRANSB) {  // B tile rows k0..k0+31, cols n0..n0+127 -> Bs[n][k]
      const int krow = tid >> 3;        // 0..31
      const int cg = (tid & 7) << 4;    // 0..112
      const float* src = Bm + (size_t)(k0 + krow) * N + n0 + cg;
#pragma unroll
      for (int i = 0; i < 16; ++i) {
        int col = n0 + cg + i;
        float v = (col < N) ? src[i] : 0.0f;
        Bs[(cg + i) * 40 + krow] = f2h(v);
      }
    } else {  // B(k,n)=Bm[n*K+k]; contiguous along k
      const int nn = tid >> 1;          // 0..127
      const int kg = (tid & 1) << 4;    // 0,16
      const int gcol = n0 + nn;
      const float* src = Bm + (size_t)gcol * K + k0 + kg;
#pragma unroll
      for (int i = 0; i < 16; ++i) {
        float v = (gcol < N) ? src[i] : 0.0f;
        Bs[nn * 40 + kg + i] = f2h(v);
      }
    }
    __syncthreads();
    if (k0 + 32 < K)  // global_prefetch_b8 of next A tile
      __builtin_prefetch(A + (size_t)(m0 + arow) * K + k0 + 32 + acol, 0, 1);

    FragH a[2], b[2];
#pragma unroll
    for (int mi = 0; mi < 2; ++mi) {
      const unsigned short* base = &As[(wm * 32 + mi * 16 + m15) * 40];
#pragma unroll
      for (int g = 0; g < 8; ++g)
        a[mi].u[g] = *(const unsigned*)&base[a_koff(g, hf)];
    }
#pragma unroll
    for (int ni = 0; ni < 2; ++ni) {
      const unsigned short* base = &Bs[(wn * 32 + ni * 16 + m15) * 40];
#pragma unroll
      for (int g = 0; g < 8; ++g)
        b[ni].u[g] = *(const unsigned*)&base[a_koff(g, hf)];
    }
#pragma unroll
    for (int mi = 0; mi < 2; ++mi)
#pragma unroll
      for (int ni = 0; ni < 2; ++ni)
        acc[mi][ni] = __builtin_amdgcn_wmma_f32_16x16x32_f16(
            false, a[mi].v, false, b[ni].v, (short)0, acc[mi][ni], false, false);
    __syncthreads();
  }

  // Epilogue. C layout: VGPR r -> row = r + 8*hf, col = lane&15.
#pragma unroll
  for (int mi = 0; mi < 2; ++mi) {
#pragma unroll
    for (int ni = 0; ni < 2; ++ni) {
      int col = n0 + wn * 32 + ni * 16 + m15;
      if (col < N) {
        float bv = bias ? bias[col] : 0.0f;
        int rbase = m0 + wm * 32 + mi * 16 + 8 * hf;
#pragma unroll
        for (int r = 0; r < 8; ++r) {
          float v = acc[mi][ni][r] + bv;
          if (GELU) v = gelu_fn(v);
          size_t idx = (size_t)(rbase + r) * N + col;
          if (RESID) v += resid[idx];
          C[idx] = v;
        }
      }
    }
  }
}

// ---------------------------------------------------------------------------
// Flash-style causal attention. Block = (64 q-rows, 1 head), 4 waves x 16 rows.
// Keys streamed in tiles of 32 via LDS. Scale folded into Q conversion.
// ---------------------------------------------------------------------------
__global__ __launch_bounds__(128) void attn_kernel(
    const float* __restrict__ qkv, const float* __restrict__ amask,
    float* __restrict__ wv) {
  __shared__ unsigned short Qs[64 * 72];        // [qrow][dim]
  __shared__ unsigned short Ks[32 * 72];        // [key][dim]
  __shared__ unsigned short Vs[64 * 40];        // [dim][key] (transposed)
  __shared__ unsigned short Ps[4 * 16 * 40];    // per-wave P scratch [row][key]

  const int tid = threadIdx.x;
  const int lane = tid & 31;
  const int wid = tid >> 5;  // 0..3
  const int hf = lane >> 4;
  const int m15 = lane & 15;
  const int head = blockIdx.y;
  const int qb = blockIdx.x;
  const int qbase = qb * 64;
  const int D3 = 3 * DD;
  const int hoff = head * HDIM;

  {  // Q: 64 rows x 64 dims, scaled by 1/sqrt(64)
    int row = tid >> 1;
    int dg = (tid & 1) * 32;
    const float* src = qkv + (size_t)(qbase + row) * D3 + hoff + dg;
#pragma unroll
    for (int i = 0; i < 32; ++i) Qs[row * 72 + dg + i] = f2h(src[i] * 0.125f);
  }
  __syncthreads();

  FragH aq[2];
#pragma unroll
  for (int c = 0; c < 2; ++c) {
    const unsigned short* base = &Qs[(wid * 16 + m15) * 72 + c * 32];
#pragma unroll
    for (int g = 0; g < 8; ++g)
      aq[c].u[g] = *(const unsigned*)&base[a_koff(g, hf)];
  }

  float mrow[8], lrow[8];
#pragma unroll
  for (int r = 0; r < 8; ++r) { mrow[r] = -__builtin_inff(); lrow[r] = 0.0f; }
  v8f o[4] = {};

  const int nsteps = 2 * (qb + 1);  // keys 0 .. qbase+63 (causal)
  for (int j = 0; j < nsteps; ++j) {
    const int kb = j * 32;
    __syncthreads();  // protect Ks/Vs from previous iteration readers
    {
      int key = tid >> 2;
      int dg = (tid & 3) * 16;
      const float* srck = qkv + (size_t)(kb + key) * D3 + DD + hoff + dg;
      const float* srcv = qkv + (size_t)(kb + key) * D3 + 2 * DD + hoff + dg;
#pragma unroll
      for (int i = 0; i < 16; ++i) Ks[key * 72 + dg + i] = f2h(srck[i]);
#pragma unroll
      for (int i = 0; i < 16; ++i) Vs[(dg + i) * 40 + key] = f2h(srcv[i]);
    }
    __syncthreads();

    // S = Q * K^T : two 16x16 tiles, each 2 chained WMMAs over dim chunks
    v8f s[2] = {};
#pragma unroll
    for (int n = 0; n < 2; ++n) {
#pragma unroll
      for (int c = 0; c < 2; ++c) {
        FragH bk;
        const unsigned short* base = &Ks[(16 * n + m15) * 72 + c * 32];
#pragma unroll
        for (int g = 0; g < 8; ++g) bk.u[g] = *(const unsigned*)&base[a_koff(g, hf)];
        s[n] = __builtin_amdgcn_wmma_f32_16x16x32_f16(
            false, aq[c].v, false, bk.v, (short)0, s[n], false, false);
      }
    }

    const int kg0 = kb + m15;
    const int kg1 = kb + 16 + m15;
    const float madd0 = (1.0f - amask[kg0]) * -3.3895314e38f;
    const float madd1 = (1.0f - amask[kg1]) * -3.3895314e38f;
#pragma unroll
    for (int r = 0; r < 8; ++r) {
      int q = qbase + wid * 16 + r + 8 * hf;
      float v0 = s[0][r] + madd0;
      float v1 = s[1][r] + madd1;
      if (kg0 > q) v0 = -__builtin_inff();
      if (kg1 > q) v1 = -__builtin_inff();
      float mx = fmaxf(v0, v1);
#pragma unroll
      for (int off = 8; off >= 1; off >>= 1)
        mx = fmaxf(mx, __shfl_xor(mx, off, 32));
      float mnew = fmaxf(mrow[r], mx);
      float corr = __expf(mrow[r] - mnew);
      float p0 = __expf(v0 - mnew);
      float p1 = __expf(v1 - mnew);
      float sum = p0 + p1;
#pragma unroll
      for (int off = 8; off >= 1; off >>= 1) sum += __shfl_xor(sum, off, 32);
      lrow[r] = lrow[r] * corr + sum;
      mrow[r] = mnew;
#pragma unroll
      for (int ni = 0; ni < 4; ++ni) o[ni][r] *= corr;
      int prow = r + 8 * hf;
      Ps[wid * 640 + prow * 40 + m15] = f2h(p0);
      Ps[wid * 640 + prow * 40 + 16 + m15] = f2h(p1);
    }
    __syncthreads();  // order Ps writes (C layout) -> Ps reads (A layout)

    FragH ap;
    {
      const unsigned short* base = &Ps[wid * 640 + m15 * 40];
#pragma unroll
      for (int g = 0; g < 8; ++g)
        ap.u[g] = *(const unsigned*)&base[a_koff(g, hf)];
    }
#pragma unroll
    for (int ni = 0; ni < 4; ++ni) {
      FragH bv;
      const unsigned short* base = &Vs[(16 * ni + m15) * 40];
#pragma unroll
      for (int g = 0; g < 8; ++g) bv.u[g] = *(const unsigned*)&base[a_koff(g, hf)];
      o[ni] = __builtin_amdgcn_wmma_f32_16x16x32_f16(
          false, ap.v, false, bv.v, (short)0, o[ni], false, false);
    }
  }

#pragma unroll
  for (int ni = 0; ni < 4; ++ni) {
    int dcol = 16 * ni + m15;
#pragma unroll
    for (int r = 0; r < 8; ++r) {
      int t = qbase + wid * 16 + r + 8 * hf;
      wv[(size_t)t * DD + hoff + dcol] = o[ni][r] / lrow[r];
    }
  }
}

// ---------------------------------------------------------------------------
__global__ __launch_bounds__(256) void ln_kernel(
    const float* __restrict__ xin, const float* __restrict__ g,
    const float* __restrict__ b, float* __restrict__ y) {
  __shared__ float rs[256], rs2[256];
  int row = blockIdx.x, tid = threadIdx.x;
  const float* xr = xin + (size_t)row * DD;
  float v[4];
  float s = 0.f, s2 = 0.f;
#pragma unroll
  for (int i = 0; i < 4; ++i) {
    v[i] = xr[tid + 256 * i];
    s += v[i];
    s2 += v[i] * v[i];
  }
  rs[tid] = s; rs2[tid] = s2;
  __syncthreads();
  for (int off = 128; off > 0; off >>= 1) {
    if (tid < off) { rs[tid] += rs[tid + off]; rs2[tid] += rs2[tid + off]; }
    __syncthreads();
  }
  float mean = rs[0] * (1.0f / DD);
  float var = rs2[0] * (1.0f / DD) - mean * mean;
  float rstd = rsqrtf(var + 1e-5f);
#pragma unroll
  for (int i = 0; i < 4; ++i) {
    int d = tid + 256 * i;
    y[(size_t)row * DD + d] = (v[i] - mean) * rstd * g[d] + b[d];
  }
}

__global__ __launch_bounds__(256) void embed_kernel(
    const int* __restrict__ x, const float* __restrict__ tok,
    const float* __restrict__ pos, float* __restrict__ hout) {
  int t = blockIdx.x, tid = threadIdx.x;
  int id = x[t];
#pragma unroll
  for (int i = 0; i < 4; ++i) {
    int d = tid + 256 * i;
    hout[(size_t)t * DD + d] = tok[(size_t)id * DD + d] + pos[(size_t)t * DD + d];
  }
}

// ---------------------------------------------------------------------------
extern "C" void kernel_launch(void* const* d_in, const int* in_sizes, int n_in,
                              void* d_out, int out_size, void* d_ws,
                              size_t ws_size, hipStream_t stream) {
  (void)in_sizes; (void)n_in; (void)out_size; (void)ws_size;
  const int* x = (const int*)d_in[0];
  const float* amask = (const float*)d_in[1];
  const float* tok_emb = (const float*)d_in[2];
  const float* pos_emb = (const float*)d_in[3];
  const float* ln1_g = (const float*)d_in[4];
  const float* ln1_b = (const float*)d_in[5];
  const float* qkv_w = (const float*)d_in[6];
  const float* qkv_b = (const float*)d_in[7];
  const float* out_w = (const float*)d_in[8];
  const float* out_b = (const float*)d_in[9];
  const float* ln2_g = (const float*)d_in[10];
  const float* ln2_b = (const float*)d_in[11];
  const float* fc1_w = (const float*)d_in[12];
  const float* fc1_b = (const float*)d_in[13];
  const float* fc2_w = (const float*)d_in[14];
  const float* fc2_b = (const float*)d_in[15];
  const float* lnf_g = (const float*)d_in[16];
  const float* lnf_b = (const float*)d_in[17];
  const float* lm_w = (const float*)d_in[18];
  float* logits = (float*)d_out;

  float* hbuf = (float*)d_ws;          // T*D
  float* xn = hbuf + TT * DD;          // T*D
  float* qkv = xn + TT * DD;           // T*3D
  float* wvb = qkv + TT * 3 * DD;      // T*D
  float* ff1 = wvb + TT * DD;          // T*4D

  embed_kernel<<<TT, 256, 0, stream>>>(x, tok_emb, pos_emb, hbuf);

  dim3 blk(256);
  for (int l = 0; l < NL; ++l) {
    ln_kernel<<<TT, 256, 0, stream>>>(hbuf, ln1_g + l * DD, ln1_b + l * DD, xn);
    gemm_kernel<false, false, false><<<dim3(3 * DD / 128, TT / 64), blk, 0, stream>>>(
        xn, qkv_w + (size_t)l * DD * 3 * DD, qkv_b + (size_t)l * 3 * DD, nullptr,
        qkv, TT, 3 * DD, DD);
    attn_kernel<<<dim3(TT / 64, NH), 128, 0, stream>>>(qkv, amask, wvb);
    gemm_kernel<false, true, false><<<dim3(DD / 128, TT / 64), blk, 0, stream>>>(
        wvb, out_w + (size_t)l * DD * DD, out_b + (size_t)l * DD, hbuf, hbuf,
        TT, DD, DD);
    ln_kernel<<<TT, 256, 0, stream>>>(hbuf, ln2_g + l * DD, ln2_b + l * DD, xn);
    gemm_kernel<false, false, true><<<dim3(4 * DD / 128, TT / 64), blk, 0, stream>>>(
        xn, fc1_w + (size_t)l * DD * 4 * DD, fc1_b + (size_t)l * 4 * DD, nullptr,
        ff1, TT, 4 * DD, DD);
    gemm_kernel<false, true, false><<<dim3(DD / 128, TT / 64), blk, 0, stream>>>(
        ff1, fc2_w + (size_t)l * 4 * DD * DD, fc2_b + (size_t)l * DD, hbuf, hbuf,
        TT, DD, 4 * DD);
  }
  ln_kernel<<<TT, 256, 0, stream>>>(hbuf, lnf_g, lnf_b, xn);
  gemm_kernel<true, false, false><<<dim3((VV + 127) / 128, TT / 64), blk, 0, stream>>>(
      xn, lm_w, nullptr, nullptr, logits, TT, VV, DD);
}
